// MultiHeadSelfAttention_69097433858554
// MI455X (gfx1250) — matrix operand order repaired and checked by reading, not compiled, not run on hardware
//
#include <hip/hip_runtime.h>

#define DEV __device__ __forceinline__

typedef __bf16 bf16_t;
typedef __attribute__((ext_vector_type(16))) __bf16 v16bf;
typedef __attribute__((ext_vector_type(8)))  __bf16 v8bf;
typedef __attribute__((ext_vector_type(8)))  float  v8f;

static constexpr int B_   = 4;
static constexpr int N_   = 2048;
static constexpr int D_   = 1024;
static constexpr int H_   = 16;
static constexpr int HD_  = 64;
static constexpr int DQKV = 3 * D_;
static constexpr int BN   = B_ * N_;

// ---- helpers -------------------------------------------------------------

DEV bf16_t f2bf(float f) {
  union { float f; unsigned u; } a; a.f = f;
  unsigned r = a.u + 0x7FFFu + ((a.u >> 16) & 1u);      // round-to-nearest-even
  union { unsigned short s; bf16_t b; } c; c.s = (unsigned short)(r >> 16);
  return c.b;
}

// B-fragment: 16 contiguous bf16 per lane (K = koff + 16*(lane>>4) + e)
DEV v16bf load_b_frag(const bf16_t* p) {
  v8bf lo = *reinterpret_cast<const v8bf*>(p);
  v8bf hi = *reinterpret_cast<const v8bf*>(p + 8);
  v16bf r;
#pragma unroll
  for (int i = 0; i < 8; ++i) { r[i] = lo[i]; r[i + 8] = hi[i]; }
  return r;
}

// A-fragment: chunk0 = p[0..7] (K=base..base+7), chunk1 = p[8..15]
// (K=base+8..base+15); base already includes 16*(lane>>4)
DEV v16bf load_a_frag(const bf16_t* p) {
  v8bf lo = *reinterpret_cast<const v8bf*>(p);
  v8bf hi = *reinterpret_cast<const v8bf*>(p + 8);
  v16bf r;
#pragma unroll
  for (int i = 0; i < 8; ++i) { r[i] = lo[i]; r[i + 8] = hi[i]; }
  return r;
}

DEV v8f wmma_bf16(v16bf a, v16bf b, v8f c) {
  return __builtin_amdgcn_wmma_f32_16x16x32_bf16(
      false, a, false, b, (short)0, c, false, false);
}

DEV v8f zero8() { v8f z; for (int i = 0; i < 8; ++i) z[i] = 0.0f; return z; }

// ---- stage 1: fp32 -> bf16 cast ------------------------------------------

__global__ void mhsa_cast_bf16(const float* __restrict__ in,
                               bf16_t* __restrict__ out, int n) {
  int i = blockIdx.x * blockDim.x + threadIdx.x;
  if (i < n) out[i] = f2bf(in[i]);
}

// ---- pipelined 32x64 GEMM core (per wave), K-loop over D in steps of 32 --
// acc[sub][j]: sub in {0,1} row sub-tiles, j in {0..3} col tiles.
// pa0/pa1: A row pointers (lane-adjusted), pb[j]: B row pointers (lane-adj).
DEV void gemm32x64(const bf16_t* pa0, const bf16_t* pa1,
                   const bf16_t* const pb[4], v8f acc[2][4]) {
  v16bf a0 = load_a_frag(pa0);
  v16bf a1 = load_a_frag(pa1);
  v16bf b0 = load_b_frag(pb[0]);
  v16bf b1 = load_b_frag(pb[1]);
  v16bf b2 = load_b_frag(pb[2]);
  v16bf b3 = load_b_frag(pb[3]);
  for (int kt = 0; kt < D_ - 32; kt += 32) {
    // issue next k-step's loads first: WMMAs below hide their latency
    v16bf a0n = load_a_frag(pa0 + kt + 32);
    v16bf a1n = load_a_frag(pa1 + kt + 32);
    v16bf b0n = load_b_frag(pb[0] + kt + 32);
    v16bf b1n = load_b_frag(pb[1] + kt + 32);
    v16bf b2n = load_b_frag(pb[2] + kt + 32);
    v16bf b3n = load_b_frag(pb[3] + kt + 32);
    acc[0][0] = wmma_bf16(a0, b0, acc[0][0]);
    acc[1][0] = wmma_bf16(a1, b0, acc[1][0]);
    acc[0][1] = wmma_bf16(a0, b1, acc[0][1]);
    acc[1][1] = wmma_bf16(a1, b1, acc[1][1]);
    acc[0][2] = wmma_bf16(a0, b2, acc[0][2]);
    acc[1][2] = wmma_bf16(a1, b2, acc[1][2]);
    acc[0][3] = wmma_bf16(a0, b3, acc[0][3]);
    acc[1][3] = wmma_bf16(a1, b3, acc[1][3]);
    a0 = a0n; a1 = a1n; b0 = b0n; b1 = b1n; b2 = b2n; b3 = b3n;
  }
  acc[0][0] = wmma_bf16(a0, b0, acc[0][0]);
  acc[1][0] = wmma_bf16(a1, b0, acc[1][0]);
  acc[0][1] = wmma_bf16(a0, b1, acc[0][1]);
  acc[1][1] = wmma_bf16(a1, b1, acc[1][1]);
  acc[0][2] = wmma_bf16(a0, b2, acc[0][2]);
  acc[1][2] = wmma_bf16(a1, b2, acc[1][2]);
  acc[0][3] = wmma_bf16(a0, b3, acc[0][3]);
  acc[1][3] = wmma_bf16(a1, b3, acc[1][3]);
}

// ---- stage 2: QKV GEMM ---------------------------------------------------

__global__ __launch_bounds__(256) void mhsa_qkv_gemm(
    const bf16_t* __restrict__ xb, const bf16_t* __restrict__ wq,
    const float* __restrict__ bias,
    bf16_t* __restrict__ qb, bf16_t* __restrict__ kb, bf16_t* __restrict__ vtb) {
  const int lane = threadIdx.x & 31;
  const int wid  = blockIdx.x * 8 + (threadIdx.x >> 5);
  const int NCB  = DQKV / 64;            // 48 column blocks
  const int colb = wid % NCB;
  const int rowb = wid / NCB;            // < BN/32 = 256
  const int n = lane & 15, h2 = lane >> 4;

  const bf16_t* pa0 = xb + (size_t)(rowb * 32 + n) * D_ + 16 * h2;
  const bf16_t* pa1 = pa0 + (size_t)16 * D_;
  const bf16_t* pb[4];
#pragma unroll
  for (int j = 0; j < 4; ++j)
    pb[j] = wq + (size_t)(colb * 64 + j * 16 + n) * D_ + 16 * h2;

  v8f acc[2][4] = {{zero8(), zero8(), zero8(), zero8()},
                   {zero8(), zero8(), zero8(), zero8()}};
  gemm32x64(pa0, pa1, pb, acc);

#pragma unroll
  for (int sub = 0; sub < 2; ++sub) {
    const int r0 = rowb * 32 + sub * 16 + 8 * h2;
#pragma unroll
    for (int j = 0; j < 4; ++j) {
      const int c = colb * 64 + j * 16 + n;
      const float bs = bias[c];
      const int which = c >> 10;         // c / D_
      const int rem = c & (D_ - 1);
      const int h = rem >> 6, hd = rem & 63;
#pragma unroll
      for (int v = 0; v < 8; ++v) {
        const int r = r0 + v;
        const int bi = r >> 11, ni = r & (N_ - 1);
        const float val = acc[sub][j][v] + bs;
        if (which == 0)
          qb[(((size_t)bi * H_ + h) * N_ + ni) * HD_ + hd] = f2bf(val * 0.125f);
        else if (which == 1)
          kb[(((size_t)bi * H_ + h) * N_ + ni) * HD_ + hd] = f2bf(val);
        else
          vtb[(((size_t)bi * H_ + h) * HD_ + hd) * N_ + ni] = f2bf(val);
      }
    }
  }
}

// ---- stage 3: flash attention per (b,h, 16-query block) ------------------

__global__ __launch_bounds__(256) void mhsa_attn(
    const bf16_t* __restrict__ q, const bf16_t* __restrict__ k,
    const bf16_t* __restrict__ vt, bf16_t* __restrict__ ctx) {
  __shared__ __align__(32) bf16_t pbuf[8][16 * 32];   // per-wave P tile
  const int lane  = threadIdx.x & 31;
  const int wslot = threadIdx.x >> 5;
  const int wid   = blockIdx.x * 8 + wslot;
  const int QB    = N_ / 16;                          // 128
  const int qbi = wid % QB, bh = wid / QB;            // bh < B*H = 64
  const int q0 = qbi * 16;
  const int n = lane & 15, h2 = lane >> 4;

  const bf16_t* qp = q  + (size_t)bh * N_ * HD_;
  const bf16_t* kp = k  + (size_t)bh * N_ * HD_;
  const bf16_t* vp = vt + (size_t)bh * HD_ * N_;

  // Q A-fragments for both Hd k-steps, resident across the KV loop
  v16bf qa0 = load_a_frag(qp + (size_t)(q0 + n) * HD_ + 0  + 16 * h2);
  v16bf qa1 = load_a_frag(qp + (size_t)(q0 + n) * HD_ + 32 + 16 * h2);

  float mrow[8], lrow[8];
  v8f acc[4] = {zero8(), zero8(), zero8(), zero8()};
#pragma unroll
  for (int v = 0; v < 8; ++v) { mrow[v] = -3.0e38f; lrow[v] = 0.0f; }

  bf16_t* myp = &pbuf[wslot][0];

  // preload K fragments for the first KV tile
  v16bf bk[2][2];
#pragma unroll
  for (int t = 0; t < 2; ++t) {
    const bf16_t* krow = kp + (size_t)(t * 16 + n) * HD_ + 16 * h2;
    bk[t][0] = load_b_frag(krow);
    bk[t][1] = load_b_frag(krow + 32);
  }

  for (int kv0 = 0; kv0 < N_; kv0 += 32) {
    // V-fragment loads first: independent of softmax, hidden under it
    v16bf bv[4];
#pragma unroll
    for (int j = 0; j < 4; ++j)
      bv[j] = load_b_frag(vp + (size_t)(j * 16 + n) * N_ + kv0 + 16 * h2);

    // scores S = Q * K^T for two 16-col tiles (Hd=64 => 2 k-steps)
    v8f s[2];
#pragma unroll
    for (int t = 0; t < 2; ++t) {
      s[t] = wmma_bf16(qa0, bk[t][0], zero8());
      s[t] = wmma_bf16(qa1, bk[t][1], s[t]);
    }

    // issue next KV tile's K fragments (hidden under softmax + PV WMMAs)
    v16bf bkn[2][2];
    if (kv0 + 32 < N_) {
#pragma unroll
      for (int t = 0; t < 2; ++t) {
        const bf16_t* krow = kp + (size_t)(kv0 + 32 + t * 16 + n) * HD_ + 16 * h2;
        bkn[t][0] = load_b_frag(krow);
        bkn[t][1] = load_b_frag(krow + 32);
      }
      __builtin_prefetch(vp + (size_t)n * N_ + kv0 + 32, 0, 0);  // global_prefetch_b8
    }

    // online softmax over the 16x32 tile; row r lives in one 16-lane half
    float alpha[8];
#pragma unroll
    for (int v = 0; v < 8; ++v) {
      float mx = fmaxf(s[0][v], s[1][v]);
      mx = fmaxf(mx, __shfl_xor(mx, 1, 32));
      mx = fmaxf(mx, __shfl_xor(mx, 2, 32));
      mx = fmaxf(mx, __shfl_xor(mx, 4, 32));
      mx = fmaxf(mx, __shfl_xor(mx, 8, 32));
      const float mnew = fmaxf(mrow[v], mx);
      alpha[v] = __expf(mrow[v] - mnew);
      mrow[v] = mnew;
      const float p0 = __expf(s[0][v] - mnew);
      const float p1 = __expf(s[1][v] - mnew);
      s[0][v] = p0; s[1][v] = p1;
      float ps = p0 + p1;
      ps += __shfl_xor(ps, 1, 32);
      ps += __shfl_xor(ps, 2, 32);
      ps += __shfl_xor(ps, 4, 32);
      ps += __shfl_xor(ps, 8, 32);
      lrow[v] = lrow[v] * alpha[v] + ps;
    }
#pragma unroll
    for (int j = 0; j < 4; ++j)
#pragma unroll
      for (int v = 0; v < 8; ++v) acc[j][v] *= alpha[v];

    // C-layout -> A-layout reshape of P through wave-private LDS
#pragma unroll
    for (int v = 0; v < 8; ++v) {
      myp[(8 * h2 + v) * 32 + n]      = f2bf(s[0][v]);
      myp[(8 * h2 + v) * 32 + 16 + n] = f2bf(s[1][v]);
    }
    asm volatile("s_wait_dscnt 0" ::: "memory");      // LDS RAW across lanes
    v16bf pa = load_a_frag(myp + n * 32 + 16 * h2);

    // O += P * V  (V stored transposed: [Hd][N] per head)
#pragma unroll
    for (int j = 0; j < 4; ++j)
      acc[j] = wmma_bf16(pa, bv[j], acc[j]);

#pragma unroll
    for (int t = 0; t < 2; ++t) { bk[t][0] = bkn[t][0]; bk[t][1] = bkn[t][1]; }
  }

  // normalize + store to ctx[B][N][H][Hd] (== [B,N,D] for the projection)
  const int bi = bh >> 4, h = bh & 15;
#pragma unroll
  for (int j = 0; j < 4; ++j)
#pragma unroll
    for (int v = 0; v < 8; ++v) {
      const int r = q0 + 8 * h2 + v;
      const float val = acc[j][v] / lrow[v];
      ctx[(((size_t)bi * N_ + r) * H_ + h) * HD_ + j * 16 + n] = f2bf(val);
    }
}

// ---- stage 4: output projection ------------------------------------------

__global__ __launch_bounds__(256) void mhsa_proj_gemm(
    const bf16_t* __restrict__ ctx, const bf16_t* __restrict__ wp,
    const float* __restrict__ bias, float* __restrict__ out) {
  const int lane = threadIdx.x & 31;
  const int wid  = blockIdx.x * 8 + (threadIdx.x >> 5);
  const int NCB  = D_ / 64;              // 16 column blocks
  const int colb = wid % NCB;
  const int rowb = wid / NCB;            // < BN/32 = 256
  const int n = lane & 15, h2 = lane >> 4;

  const bf16_t* pa0 = ctx + (size_t)(rowb * 32 + n) * D_ + 16 * h2;
  const bf16_t* pa1 = pa0 + (size_t)16 * D_;
  const bf16_t* pb[4];
#pragma unroll
  for (int j = 0; j < 4; ++j)
    pb[j] = wp + (size_t)(colb * 64 + j * 16 + n) * D_ + 16 * h2;

  v8f acc[2][4] = {{zero8(), zero8(), zero8(), zero8()},
                   {zero8(), zero8(), zero8(), zero8()}};
  gemm32x64(pa0, pa1, pb, acc);

#pragma unroll
  for (int sub = 0; sub < 2; ++sub) {
    const int r0 = rowb * 32 + sub * 16 + 8 * h2;
#pragma unroll
    for (int j = 0; j < 4; ++j) {
      const int c = colb * 64 + j * 16 + n;
      const float bs = bias[c];
#pragma unroll
      for (int v = 0; v < 8; ++v)
        out[(size_t)(r0 + v) * D_ + c] = acc[sub][j][v] + bs;
    }
  }
}

// ---- launcher ------------------------------------------------------------

extern "C" void kernel_launch(void* const* d_in, const int* in_sizes, int n_in,
                              void* d_out, int out_size, void* d_ws, size_t ws_size,
                              hipStream_t stream) {
  (void)in_sizes; (void)n_in; (void)out_size; (void)ws_size;
  const float* x      = (const float*)d_in[0];
  const float* qkv_w  = (const float*)d_in[1];
  const float* qkv_b  = (const float*)d_in[2];
  const float* proj_w = (const float*)d_in[3];
  const float* proj_b = (const float*)d_in[4];
  float* out = (float*)d_out;

  char* ws = (char*)d_ws;
  size_t off = 0;
  bf16_t* xb  = (bf16_t*)(ws + off); off += (size_t)BN * D_ * 2;     // 16 MB
  bf16_t* wq  = (bf16_t*)(ws + off); off += (size_t)DQKV * D_ * 2;   //  6 MB
  bf16_t* wp  = (bf16_t*)(ws + off); off += (size_t)D_ * D_ * 2;     //  2 MB
  bf16_t* qb  = (bf16_t*)(ws + off); off += (size_t)BN * D_ * 2;     // 16 MB
  bf16_t* kb  = (bf16_t*)(ws + off); off += (size_t)BN * D_ * 2;     // 16 MB
  bf16_t* vtb = (bf16_t*)(ws + off); off += (size_t)BN * D_ * 2;     // 16 MB
  bf16_t* ctx = (bf16_t*)(ws + off); off += (size_t)BN * D_ * 2;     // 16 MB

  const int T = 256;
  mhsa_cast_bf16<<<(BN * D_ + T - 1) / T, T, 0, stream>>>(x, xb, BN * D_);
  mhsa_cast_bf16<<<(DQKV * D_ + T - 1) / T, T, 0, stream>>>(qkv_w, wq, DQKV * D_);
  mhsa_cast_bf16<<<(D_ * D_ + T - 1) / T, T, 0, stream>>>(proj_w, wp, D_ * D_);

  // QKV: (BN/32)=256 row blocks x (3D/64)=48 col blocks = 12288 waves
  mhsa_qkv_gemm<<<12288 / 8, T, 0, stream>>>(xb, wq, qkv_b, qb, kb, vtb);
  // attention: B*H*(N/16) = 8192 waves
  mhsa_attn<<<8192 / 8, T, 0, stream>>>(qb, kb, vtb, ctx);
  // projection: 256 x 16 = 4096 waves
  mhsa_proj_gemm<<<4096 / 8, T, 0, stream>>>(ctx, wp, proj_b, out);
}
